// FastJLWrapper_2491081032317
// MI455X (gfx1250) — matrix operand order repaired and checked by reading, not compiled, not run on hardware
//
#include <hip/hip_runtime.h>
#include <hip/hip_bf16.h>

typedef _Float16 v4h  __attribute__((ext_vector_type(4)));
typedef _Float16 v8h  __attribute__((ext_vector_type(8)));
typedef _Float16 v16h __attribute__((ext_vector_type(16)));
typedef float    v8f  __attribute__((ext_vector_type(8)));

#define MDIM 512
#define KDIM 4096
#define NDIM 4096

// LDS staging: K-slab of 64, rows padded to 72 f16 (144B) => ds_load_b128
// start banks are distinct across lanes (36 words mod 64). Dynamic shared
// memory => LDS base offset 0, so async byte offsets == pointer offsets.
#define ROWLEN  72
#define A_BUF   (64  * ROWLEN)          // 4608 f16
#define B_BUF   (256 * ROWLEN)          // 18432 f16
#define A0F     0u                       // f16 indices
#define A1F     (A_BUF)
#define B0F     (2u * A_BUF)
#define B1F     (2u * A_BUF + B_BUF)
#define A0B     0u                       // byte offsets
#define A1B     (2u * A_BUF)
#define B0B     (4u * A_BUF)
#define B1B     (4u * A_BUF + 2u * B_BUF)
#define LDS_F16_TOTAL (2 * A_BUF + 2 * B_BUF)   // 46080 f16 = 92160 B

// ---------------------------------------------------------------------------
// 1024-point FWHT of D[:1024]*V for each of the 4 params (one block each).
// ---------------------------------------------------------------------------
__global__ void fwht1024_x4(const float* __restrict__ V,
                            const float* __restrict__ D0, const float* __restrict__ D1,
                            const float* __restrict__ D2, const float* __restrict__ D3,
                            float* __restrict__ wout)
{
    __shared__ float s[1024];
    const float* Dp = (blockIdx.x == 0) ? D0 : (blockIdx.x == 1) ? D1
                    : (blockIdx.x == 2) ? D2 : D3;
    const int t = threadIdx.x;              // 512 threads
    s[t]       = Dp[t]       * V[t];
    s[t + 512] = Dp[t + 512] * V[t + 512];
    __syncthreads();
    #pragma unroll
    for (int st = 512; st >= 1; st >>= 1) {
        const int j = ((t & ~(st - 1)) << 1) | (t & (st - 1));
        const float a = s[j], b = s[j + st];
        s[j] = a + b;
        s[j + st] = a - b;
        __syncthreads();
    }
    float* o = wout + (size_t)blockIdx.x * 1024;
    o[t]       = s[t];
    o[t + 512] = s[t + 512];
}

// ---------------------------------------------------------------------------
// Wh[flat] = f16( W_init[flat] + PP[flat] * w[flat & 1023] / 1024 )
// ---------------------------------------------------------------------------
__global__ void materialize_w(const float4* __restrict__ Wi,
                              const float4* __restrict__ PP,
                              const float*  __restrict__ w,
                              v4h* __restrict__ Wh)
{
    const int i = blockIdx.x * blockDim.x + threadIdx.x;   // float4 index
    const float4 a = Wi[i];
    const float4 p = PP[i];
    const float4 wv = *(const float4*)(w + ((i << 2) & 1023));
    const float sc = 1.0f / 1024.0f;
    v4h o;
    o[0] = (_Float16)(a.x + p.x * wv.x * sc);
    o[1] = (_Float16)(a.y + p.y * wv.y * sc);
    o[2] = (_Float16)(a.z + p.z * wv.z * sc);
    o[3] = (_Float16)(a.w + p.w * wv.w * sc);
    Wh[i] = o;
}

__global__ void convert_f32_f16(const float4* __restrict__ in, v4h* __restrict__ out)
{
    const int i = blockIdx.x * blockDim.x + threadIdx.x;
    const float4 a = in[i];
    v4h o;
    o[0] = (_Float16)a.x; o[1] = (_Float16)a.y;
    o[2] = (_Float16)a.z; o[3] = (_Float16)a.w;
    out[i] = o;
}

__global__ void make_bias(const float* __restrict__ bi, const float* __restrict__ pp,
                          const float* __restrict__ w,  float* __restrict__ out)
{
    const int i = blockIdx.x * blockDim.x + threadIdx.x;   // 4096
    out[i] = bi[i] + pp[i] * w[i & 1023] * (1.0f / 1024.0f);
}

// ---------------------------------------------------------------------------
// Async copy helpers (gfx1250): 16B per lane, global -> LDS, ASYNCcnt-tracked.
// ---------------------------------------------------------------------------
__device__ __forceinline__ void async16(unsigned lds_byte, const void* g)
{
    asm volatile("global_load_async_to_lds_b128 %0, %1, off"
                 :: "v"(lds_byte), "v"(g)
                 : "memory");
}
#define WAIT_ASYNC_10() asm volatile("s_wait_asynccnt 0xa" ::: "memory")
#define WAIT_ASYNC_0()  asm volatile("s_wait_asynccnt 0x0" ::: "memory")

__device__ __forceinline__ v16h cat8(v8h lo, v8h hi)
{
    return __builtin_shufflevector(lo, hi, 0,1,2,3,4,5,6,7,8,9,10,11,12,13,14,15);
}

// Consume one staged 64-wide K-slab: 24 ds_load_b128 + 16 wmma per wave.
// Fragment layouts per CDNA5 ISA 7.12.2 (A 16-bit 16x32; B column-per-lane).
__device__ __forceinline__ void consume_tile(const _Float16* __restrict__ smem,
                                             unsigned aBase, unsigned bBase,
                                             int wm, int wn, int half, int l15,
                                             v8f (&acc)[2][4])
{
    #pragma unroll
    for (int ks = 0; ks < 64; ks += 32) {
        v16h af[2], bf[4];
        #pragma unroll
        for (int mt = 0; mt < 2; ++mt) {
            const unsigned base = aBase +
                (unsigned)((wm * 32 + mt * 16 + l15) * ROWLEN + ks + half * 8);
            const v8h lo = *(const v8h*)(smem + base);
            const v8h hi = *(const v8h*)(smem + base + 16);
            af[mt] = cat8(lo, hi);
        }
        #pragma unroll
        for (int nt = 0; nt < 4; ++nt) {
            const unsigned base = bBase +
                (unsigned)((wn * 64 + nt * 16 + l15) * ROWLEN + ks + half * 8);
            const v8h lo = *(const v8h*)(smem + base);
            const v8h hi = *(const v8h*)(smem + base + 16);
            bf[nt] = cat8(lo, hi);
        }
        #pragma unroll
        for (int mt = 0; mt < 2; ++mt)
            #pragma unroll
            for (int nt = 0; nt < 4; ++nt)
                acc[mt][nt] = __builtin_amdgcn_wmma_f32_16x16x32_f16(
                    false, af[mt], false, bf[nt], (short)0, acc[mt][nt], false, false);
    }
}

// ---------------------------------------------------------------------------
// GEMM: Out[m][n] = act( sum_k A[m][k] * Bw[n][k] + bias[n] )
// Block = 256 threads (8 waves, 2x4), block tile 64(M) x 256(N), K-slab 64,
// double-buffered LDS staging via global_load_async_to_lds_b128.
// ---------------------------------------------------------------------------
template<bool RELU, bool OUT16>
__global__ __launch_bounds__(256) void gemm_bias(
    const _Float16* __restrict__ A,
    const _Float16* __restrict__ Bw,
    const float*    __restrict__ bias,
    void*           __restrict__ Out)
{
    extern __shared__ _Float16 smem[];

    const int tid  = threadIdx.x;
    const int wave = tid >> 5, lane = tid & 31;
    const int half = lane >> 4, l15 = lane & 15;
    const int wm = wave >> 2, wn = wave & 3;
    const int m_blk = blockIdx.y * 64;
    const int n_blk = blockIdx.x * 256;
    const int m_w = m_blk + wm * 32;
    const int n_w = n_blk + wn * 64;

    v8f acc[2][4] = {};

    // Per-thread fill assignment: 64-wide slab = (64+256) rows x 8 chunks of
    // 16B = 2560 chunks = 10 per thread (2 A rows + 8 B rows), one 16B seg each.
    const int rb = tid >> 3;                    // 0..31
    const int sg = (tid & 7) * 8;               // f16 offset of 16B seg in row
    // Loop-carried global pointers, advanced +64 f16 (128B) per fill.
    const _Float16* ga0 = A  + (size_t)(m_blk + rb) * KDIM + sg;
    const _Float16* ga1 = ga0 + (size_t)32 * KDIM;
    const _Float16* gb[8];
    #pragma unroll
    for (int j = 0; j < 8; ++j)
        gb[j] = Bw + (size_t)(n_blk + rb + 32 * j) * KDIM + sg;
    // Static LDS byte offsets within a buffer.
    const unsigned aL0 = (unsigned)(rb * ROWLEN + sg) * 2u;
    const unsigned aL1 = aL0 + 32u * ROWLEN * 2u;
    unsigned bL[8];
    #pragma unroll
    for (int j = 0; j < 8; ++j)
        bL[j] = (unsigned)((rb + 32 * j) * ROWLEN + sg) * 2u;

    auto fill = [&](unsigned aBuf, unsigned bBuf) {
        async16(aBuf + aL0, ga0);  ga0 += 64;
        async16(aBuf + aL1, ga1);  ga1 += 64;
        #pragma unroll
        for (int j = 0; j < 8; ++j) {
            async16(bBuf + bL[j], gb[j]);
            gb[j] += 64;
        }
    };

    fill(A0B, B0B);                             // slab #0 -> buf0

    #pragma unroll 1
    for (int it = 0; it < KDIM / 128; ++it) {   // 32 iterations, 2 slabs each
        fill(A1B, B1B);                         // slab 2it+1 -> buf1
        WAIT_ASYNC_10();                        // slab 2it resident
        __syncthreads();
        consume_tile(smem, A0F, B0F, wm, wn, half, l15, acc);
        __syncthreads();                        // buf0 free

        if (it < KDIM / 128 - 1) {
            fill(A0B, B0B);                     // slab 2it+2 -> buf0
            WAIT_ASYNC_10();                    // slab 2it+1 resident
        } else {
            WAIT_ASYNC_0();
        }
        __syncthreads();
        consume_tile(smem, A1F, B1F, wm, wn, half, l15, acc);
        __syncthreads();                        // buf1 free
    }

    // Epilogue. D layout: VGPR r, lanes 0-15 -> (M = r,   N = lane)
    //                            lanes 16-31 -> (M = r+8, N = lane-16)
    #pragma unroll
    for (int mt = 0; mt < 2; ++mt) {
        #pragma unroll
        for (int nt = 0; nt < 4; ++nt) {
            const int n = n_w + nt * 16 + l15;
            const float bv = bias[n];
            const int mbase = m_w + mt * 16 + half * 8;
            #pragma unroll
            for (int r = 0; r < 8; ++r) {
                float v = acc[mt][nt][r] + bv;
                if (RELU) v = v > 0.0f ? v : 0.0f;
                if (OUT16)
                    ((_Float16*)Out)[(size_t)(mbase + r) * NDIM + n] = (_Float16)v;
                else
                    ((float*)Out)[(size_t)(mbase + r) * NDIM + n] = v;
            }
        }
    }
}

// ---------------------------------------------------------------------------
extern "C" void kernel_launch(void* const* d_in, const int* in_sizes, int n_in,
                              void* d_out, int out_size, void* d_ws, size_t ws_size,
                              hipStream_t stream)
{
    (void)in_sizes; (void)n_in; (void)out_size; (void)ws_size;

    const float* x    = (const float*)d_in[0];
    const float* V    = (const float*)d_in[1];
    const float* W1i  = (const float*)d_in[2];
    const float* DW1  = (const float*)d_in[3];
    const float* PPW1 = (const float*)d_in[4];
    const float* b1i  = (const float*)d_in[5];
    const float* Db1  = (const float*)d_in[6];
    const float* PPb1 = (const float*)d_in[7];
    const float* W2i  = (const float*)d_in[8];
    const float* DW2  = (const float*)d_in[9];
    const float* PPW2 = (const float*)d_in[10];
    const float* b2i  = (const float*)d_in[11];
    const float* Db2  = (const float*)d_in[12];
    const float* PPb2 = (const float*)d_in[13];

    char* ws = (char*)d_ws;
    _Float16* W1h = (_Float16*)(ws);                          // 32 MB
    _Float16* W2h = (_Float16*)(ws + ((size_t)32 << 20));     // 32 MB
    _Float16* xh  = (_Float16*)(ws + ((size_t)64 << 20));     //  4 MB
    _Float16* hh  = (_Float16*)(ws + ((size_t)68 << 20));     //  4 MB
    float*    w4  = (float*)   (ws + ((size_t)72 << 20));     // 4x1024 f32
    float*    b1f = w4 + 4096;
    float*    b2f = b1f + 4096;

    // 1) tiny FWHTs:  w4 = [w_W1 | w_b1 | w_W2 | w_b2]
    fwht1024_x4<<<4, 512, 0, stream>>>(V, DW1, Db1, DW2, Db2, w4);

    // 2) materialize f16 weights + biases + f16 activations
    const int n4w = (KDIM * NDIM) / 4;
    materialize_w<<<n4w / 256, 256, 0, stream>>>((const float4*)W1i, (const float4*)PPW1, w4,        (v4h*)W1h);
    materialize_w<<<n4w / 256, 256, 0, stream>>>((const float4*)W2i, (const float4*)PPW2, w4 + 2048, (v4h*)W2h);

    const int n4x = (MDIM * KDIM) / 4;
    convert_f32_f16<<<n4x / 256, 256, 0, stream>>>((const float4*)x, (v4h*)xh);

    make_bias<<<4096 / 256, 256, 0, stream>>>(b1i, PPb1, w4 + 1024, b1f);
    make_bias<<<4096 / 256, 256, 0, stream>>>(b2i, PPb2, w4 + 3072, b2f);

    // 3) GEMM chain: h = relu(x @ W1^T + b1) ; out = h @ W2^T + b2
    const size_t smem_bytes = (size_t)LDS_F16_TOTAL * sizeof(_Float16);   // 92160 B
    dim3 grid(NDIM / 256, MDIM / 64);                                     // (16, 8)
    gemm_bias<true,  true ><<<grid, 256, smem_bytes, stream>>>(xh, W1h, b1f, (void*)hh);
    gemm_bias<false, false><<<grid, 256, smem_bytes, stream>>>(hh, W2h, b2f, d_out);
}
